// InteractionBlock_73495480369395
// MI455X (gfx1250) — matrix operand order repaired and checked by reading, not compiled, not run on hardware
//
#include <hip/hip_runtime.h>
#include <hip/hip_bf16.h>
#include <math.h>

typedef _Float16 half_t;
typedef __attribute__((ext_vector_type(16))) _Float16 v16h;
typedef __attribute__((ext_vector_type(8)))  _Float16 v8h;
typedef __attribute__((ext_vector_type(4)))  _Float16 v4h;
typedef __attribute__((ext_vector_type(2)))  _Float16 v2h;
typedef __attribute__((ext_vector_type(8)))  float    v8f;

#define HID 128
#define NF  128
#define NG  50
#define NGP 64   // NG padded to multiple of 32 for K-tiling

// ---------- math helpers ----------
__device__ __forceinline__ float sspf(float x) {
    // shifted softplus: log(1+exp(x)) - log(2)
    float sp = (x > 20.0f) ? x : log1pf(expf(x));
    return sp - 0.69314718055994530942f;
}

// ---------- WMMA fragment loaders (wave32, 16x16x32 f16) ----------
// A-matrix 16x32 (MxK), f16: lane holds row m=l&15.
//   lanes 0-15 : halves K = k0+0..7   and k0+16..23
//   lanes 16-31: halves K = k0+8..15  and k0+24..31
__device__ __forceinline__ v16h load_a(const half_t* row, int k0, int hi) {
    const int koff = k0 + (hi ? 8 : 0);
    v8h lo = *(const v8h*)(row + koff);
    v8h up = *(const v8h*)(row + koff + 16);
    v16h a;
#pragma unroll
    for (int i = 0; i < 8; ++i) { a[i] = lo[i]; a[i + 8] = up[i]; }
    return a;
}

// B-matrix 32x16 (KxN), f16: lane holds column n=l&15 (weight row n, since B=W^T).
//   lanes 0-15 hold K = k0+0..7 and k0+16..23, lanes 16-31 hold K = k0+8..15 and k0+24..31
__device__ __forceinline__ v16h load_b(const half_t* row, int k0, int hi) {
    const int koff = k0 + (hi ? 8 : 0);
    v8h lo = *(const v8h*)(row + koff);
    v8h up = *(const v8h*)(row + koff + 16);
    v16h b;
#pragma unroll
    for (int i = 0; i < 8; ++i) { b[i] = lo[i]; b[i + 8] = up[i]; }
    return b;
}

__device__ __forceinline__ v8f wmma16(v16h a, v16h b, v8f c) {
    return __builtin_amdgcn_wmma_f32_16x16x32_f16(false, a, false, b, (short)0, c, false, false);
}

__device__ __forceinline__ v4h cvt4(float4 v) {
    v4h h;
    h[0] = (half_t)v.x; h[1] = (half_t)v.y; h[2] = (half_t)v.z; h[3] = (half_t)v.w;
    return h;
}

// ---------- kernel 0: zero agg, convert weights to f16 ----------
__global__ void prep_kernel(const float* __restrict__ fw1, const float4* __restrict__ fw2,
                            const float4* __restrict__ w1,  const float4* __restrict__ w2,
                            const float4* __restrict__ w3,
                            half_t* __restrict__ fw1h, half_t* __restrict__ fw2h,
                            half_t* __restrict__ w1h,  half_t* __restrict__ w2h,
                            half_t* __restrict__ w3h,  float4* __restrict__ agg4, int N) {
    const int i = blockIdx.x * blockDim.x + threadIdx.x;
    const int stride = gridDim.x * blockDim.x;
    for (int j = i; j < NF * NGP; j += stride) {
        int r = j >> 6, c = j & 63;
        fw1h[j] = (half_t)((c < NG) ? fw1[r * NG + c] : 0.0f);
    }
    for (int j = i; j < (NF * NF) / 4; j += stride) {
        *(v4h*)(fw2h + j * 4) = cvt4(fw2[j]);
        *(v4h*)(w1h  + j * 4) = cvt4(w1[j]);
        *(v4h*)(w2h  + j * 4) = cvt4(w2[j]);
        *(v4h*)(w3h  + j * 4) = cvt4(w3[j]);
    }
    const int tot4 = N * (NF / 4);
    const float4 z = make_float4(0.f, 0.f, 0.f, 0.f);
    for (int j = i; j < tot4; j += stride) agg4[j] = z;
}

// ---------- kernel 1: xh = x @ w1^T  [N,128] ----------
__global__ __launch_bounds__(32) void xh_kernel(const float* __restrict__ x,
                                                const half_t* __restrict__ w1h,
                                                float* __restrict__ xh, int N) {
    __shared__ __align__(16) half_t xa[16 * HID];
    const int lane = threadIdx.x & 31;
    const int hi = lane >> 4;
    const int mn = lane & 15;
    const int base = blockIdx.x * 16;
    const bool full = (base + 16 <= N);

    if (full) {
        const float4* x4 = (const float4*)(x + base * HID);
        for (int i = lane; i < (16 * HID) / 4; i += 32)
            *(v4h*)(xa + i * 4) = cvt4(x4[i]);
    } else {
        for (int i = lane; i < 16 * HID; i += 32) {
            int r = i >> 7, c = i & 127;
            int g = base + r;
            xa[i] = (half_t)((g < N) ? x[g * HID + c] : 0.0f);
        }
    }
    __syncthreads();

    v16h a[4];
#pragma unroll
    for (int kt = 0; kt < 4; ++kt) a[kt] = load_a(xa + mn * HID, kt * 32, hi);

#pragma unroll
    for (int nt = 0; nt < 8; ++nt) {
        const int n = nt * 16 + mn;
        v8f acc = {};
#pragma unroll
        for (int kt = 0; kt < 4; ++kt)
            acc = wmma16(a[kt], load_b(w1h + n * HID, kt * 32, hi), acc);
        if (full) {
#pragma unroll
            for (int r = 0; r < 8; ++r)
                xh[(base + r + hi * 8) * NF + n] = acc[r];
        } else {
#pragma unroll
            for (int r = 0; r < 8; ++r) {
                int g = base + r + hi * 8;
                if (g < N) xh[g * NF + n] = acc[r];
            }
        }
    }
}

// ---------- kernel 2: fused edge pipeline ----------
// h = ssp(basis@fw1^T + fb1); W = (h@fw2^T + fb2)*C; atomic scatter xh[src]*W -> agg[dst]
__global__ __launch_bounds__(32) void edge_kernel(
    const int* __restrict__ pairs, const float* __restrict__ e_ji,
    const float* __restrict__ basis,
    const half_t* __restrict__ fw1h, const float* __restrict__ fb1,
    const half_t* __restrict__ fw2h, const float* __restrict__ fb2,
    const float* __restrict__ xh, float* __restrict__ agg, int E) {
    __shared__ __align__(16) half_t bsh[16 * NGP];
    __shared__ __align__(16) half_t hsh[16 * NF];
    __shared__ float  ccut[16];
    __shared__ int    srcs[16];
    __shared__ int    dsts[16];

    const int lane = threadIdx.x & 31;
    const int hi = lane >> 4;
    const int mn = lane & 15;
    const int e0 = blockIdx.x * 16;
    const bool full = (e0 + 16 <= E);

    if (lane < 16) {
        int e = e0 + lane;
        if (e < E) {
            // cosine cutoff: 0.25*(cos(r*pi/10)+1)
            ccut[lane] = 0.25f * (cosf(e_ji[e] * 0.31415926535897932f) + 1.0f);
            srcs[lane] = pairs[e];
            dsts[lane] = pairs[E + e];
        } else {
            ccut[lane] = 0.0f; srcs[lane] = 0; dsts[lane] = 0;
        }
    }
    if (full) {
        // NG==50 is even -> float2 pairs never straddle the pad boundary
        for (int i = lane; i < (16 * NGP) / 2; i += 32) {
            int r = i >> 5;            // 32 half-pairs per padded row
            int c2 = (i & 31) * 2;
            v2h h2;
            if (c2 < NG) {
                float2 v = *(const float2*)(basis + (e0 + r) * NG + c2);
                h2[0] = (half_t)v.x; h2[1] = (half_t)v.y;
            } else {
                h2[0] = (half_t)0.0f; h2[1] = (half_t)0.0f;
            }
            *(v2h*)(bsh + i * 2) = h2;
        }
    } else {
        for (int i = lane; i < 16 * NGP; i += 32) {
            int r = i >> 6, c = i & 63;
            int e = e0 + r;
            bsh[i] = (half_t)((c < NG && e < E) ? basis[e * NG + c] : 0.0f);
        }
    }
    __syncthreads();

    // ---- layer 1: h = ssp(basis @ fw1^T + fb1) -> LDS (f16) ----
    v16h a0 = load_a(bsh + mn * NGP, 0, hi);
    v16h a1 = load_a(bsh + mn * NGP, 32, hi);
#pragma unroll
    for (int nt = 0; nt < 8; ++nt) {
        const int n = nt * 16 + mn;
        v8f acc = {};
        acc = wmma16(a0, load_b(fw1h + n * NGP, 0, hi), acc);
        acc = wmma16(a1, load_b(fw1h + n * NGP, 32, hi), acc);
        const float bias = fb1[n];
#pragma unroll
        for (int r = 0; r < 8; ++r)
            hsh[(r + hi * 8) * NF + n] = (half_t)sspf(acc[r] + bias);
    }
    __syncthreads();

    // ---- layer 2: W = (h @ fw2^T + fb2) * C, then gather/scatter ----
    v16h ha[4];
#pragma unroll
    for (int kt = 0; kt < 4; ++kt) ha[kt] = load_a(hsh + mn * NF, kt * 32, hi);

    // cache per-row metadata in registers (read LDS once)
    int   sOff[8];
    int   dOff[8];
    float cc[8];
#pragma unroll
    for (int r = 0; r < 8; ++r) {
        const int m = r + hi * 8;
        sOff[r] = srcs[m] * NF;
        dOff[r] = dsts[m] * NF;
        cc[r]   = ccut[m];
    }

#pragma unroll
    for (int nt = 0; nt < 8; ++nt) {
        const int n = nt * 16 + mn;
        v8f acc = {};
#pragma unroll
        for (int kt = 0; kt < 4; ++kt)
            acc = wmma16(ha[kt], load_b(fw2h + n * NF, kt * 32, hi), acc);
        const float bias = fb2[n];
        if (full) {
#pragma unroll
            for (int r = 0; r < 8; ++r) {
                float wv = (acc[r] + bias) * cc[r];
                float xv = xh[sOff[r] + n];
                atomicAdd(&agg[dOff[r] + n], wv * xv);
            }
        } else {
#pragma unroll
            for (int r = 0; r < 8; ++r) {
                if (e0 + r + hi * 8 < E) {
                    float wv = (acc[r] + bias) * cc[r];
                    float xv = xh[sOff[r] + n];
                    atomicAdd(&agg[dOff[r] + n], wv * xv);
                }
            }
        }
    }
}

// ---------- kernel 3: out = ssp(agg@w2^T + b2) @ w3^T + b3 ----------
__global__ __launch_bounds__(32) void out_kernel(
    const float* __restrict__ agg,
    const half_t* __restrict__ w2h, const float* __restrict__ b2,
    const half_t* __restrict__ w3h, const float* __restrict__ b3,
    float* __restrict__ out, int N) {
    __shared__ __align__(16) half_t ta[16 * NF];
    __shared__ __align__(16) half_t tb[16 * HID];
    const int lane = threadIdx.x & 31;
    const int hi = lane >> 4;
    const int mn = lane & 15;
    const int base = blockIdx.x * 16;
    const bool full = (base + 16 <= N);

    if (full) {
        const float4* a4 = (const float4*)(agg + base * NF);
        for (int i = lane; i < (16 * NF) / 4; i += 32)
            *(v4h*)(ta + i * 4) = cvt4(a4[i]);
    } else {
        for (int i = lane; i < 16 * NF; i += 32) {
            int r = i >> 7, c = i & 127;
            int g = base + r;
            ta[i] = (half_t)((g < N) ? agg[g * NF + c] : 0.0f);
        }
    }
    __syncthreads();

    v16h a[4];
#pragma unroll
    for (int kt = 0; kt < 4; ++kt) a[kt] = load_a(ta + mn * NF, kt * 32, hi);

#pragma unroll
    for (int nt = 0; nt < 8; ++nt) {
        const int n = nt * 16 + mn;
        v8f acc = {};
#pragma unroll
        for (int kt = 0; kt < 4; ++kt)
            acc = wmma16(a[kt], load_b(w2h + n * NF, kt * 32, hi), acc);
        const float bias = b2[n];
#pragma unroll
        for (int r = 0; r < 8; ++r)
            tb[(r + hi * 8) * HID + n] = (half_t)sspf(acc[r] + bias);
    }
    __syncthreads();

    v16h c4[4];
#pragma unroll
    for (int kt = 0; kt < 4; ++kt) c4[kt] = load_a(tb + mn * HID, kt * 32, hi);

#pragma unroll
    for (int nt = 0; nt < 8; ++nt) {
        const int n = nt * 16 + mn;
        v8f acc = {};
#pragma unroll
        for (int kt = 0; kt < 4; ++kt)
            acc = wmma16(c4[kt], load_b(w3h + n * HID, kt * 32, hi), acc);
        const float bias = b3[n];
        if (full) {
#pragma unroll
            for (int r = 0; r < 8; ++r)
                out[(base + r + hi * 8) * HID + n] = acc[r] + bias;
        } else {
#pragma unroll
            for (int r = 0; r < 8; ++r) {
                int g = base + r + hi * 8;
                if (g < N) out[g * HID + n] = acc[r] + bias;
            }
        }
    }
}

extern "C" void kernel_launch(void* const* d_in, const int* in_sizes, int n_in,
                              void* d_out, int out_size, void* d_ws, size_t ws_size,
                              hipStream_t stream) {
    const float* x     = (const float*)d_in[0];
    const int*   pairs = (const int*)  d_in[1];   // [2,E] src row then dst row
    const float* e_ji  = (const float*)d_in[2];
    const float* basis = (const float*)d_in[3];
    const float* fw1   = (const float*)d_in[4];
    const float* fb1   = (const float*)d_in[5];
    const float* fw2   = (const float*)d_in[6];
    const float* fb2   = (const float*)d_in[7];
    const float* w1    = (const float*)d_in[8];
    const float* w2    = (const float*)d_in[9];
    const float* b2    = (const float*)d_in[10];
    const float* w3    = (const float*)d_in[11];
    const float* b3    = (const float*)d_in[12];

    const int N = in_sizes[0] / HID;
    const int E = in_sizes[2];

    char* ws = (char*)d_ws;
    size_t off = 0;
    float* agg = (float*)(ws + off); off += (size_t)N * NF * sizeof(float);
    float* xh  = (float*)(ws + off); off += (size_t)N * NF * sizeof(float);
    half_t* fw1h = (half_t*)(ws + off); off += (size_t)NF * NGP * sizeof(half_t);
    half_t* fw2h = (half_t*)(ws + off); off += (size_t)NF * NF  * sizeof(half_t);
    half_t* w1h  = (half_t*)(ws + off); off += (size_t)NF * HID * sizeof(half_t);
    half_t* w2h  = (half_t*)(ws + off); off += (size_t)HID * NF * sizeof(half_t);
    half_t* w3h  = (half_t*)(ws + off); off += (size_t)HID * HID * sizeof(half_t);

    prep_kernel<<<2048, 256, 0, stream>>>(fw1, (const float4*)fw2,
                                          (const float4*)w1, (const float4*)w2,
                                          (const float4*)w3,
                                          fw1h, fw2h, w1h, w2h, w3h,
                                          (float4*)agg, N);
    xh_kernel<<<(N + 15) / 16, 32, 0, stream>>>(x, w1h, xh, N);
    edge_kernel<<<(E + 15) / 16, 32, 0, stream>>>(pairs, e_ji, basis,
                                                  fw1h, fb1, fw2h, fb2, xh, agg, E);
    out_kernel<<<(N + 15) / 16, 32, 0, stream>>>(agg, w2h, b2, w3h, b3,
                                                 (float*)d_out, N);
}
